// TemporalDifferenceMambaBlock_21775484190907
// MI455X (gfx1250) — matrix operand order repaired and checked by reading, not compiled, not run on hardware
//
#include <hip/hip_runtime.h>
#include <math.h>

// ---------------- problem constants ----------------
#define CNUM   64
#define TNUM   160
#define HNUM   8
#define WNUM   8
#define BNUM   2
#define LSEQ   10240            // T*H*W
#define BL     20480            // B*L
#define DI     128
#define DSN    16
#define DRN    4
#define CHK    128              // scan chunk length
#define NCH    80               // LSEQ / CHK
#define PAIRS  2048             // DI*DSN
#define EPSV   1e-5f
#define THETA  0.5f

typedef _Float16 v16h __attribute__((ext_vector_type(16)));
typedef float    v8f  __attribute__((ext_vector_type(8)));

// =====================================================================
// WMMA GEMM:  D[m][n] = sum_k A[m][k] * W[n][k]   (A row-major, W row-major NxK)
// fp32 in memory, fp16 WMMA compute, fp32 accumulate.
// Each wave computes a 16(M) x 32(N) tile: one A fragment feeds two
// back-to-back v_wmma with independent accumulators (XDL pipelining).
// K is a template constant -> fully unrolled K loop, multiple wmma instrs.
// grid.x = M/64 (4 waves/block, one M-tile each), grid.y = ceil(N/32).
// =====================================================================
template<int K>
__global__ __launch_bounds__(128)
void wmma_gemm_nt(const float* __restrict__ A, int lda,
                  const float* __restrict__ W, int ldw,
                  float* __restrict__ D, int ldd, int Nact) {
  const int lane = threadIdx.x & 31;
  const int wv   = threadIdx.x >> 5;
  const int hi   = lane >> 4;            // 0 or 1
  const long m0  = (long)(blockIdx.x * 4 + wv) * 16;
  const int  n0  = blockIdx.y * 32;
  const long mrow = m0 + (lane & 15);
  const int  n1   = n0 + (lane & 15);
  const int  n2   = n1 + 16;
  const int  kA   = hi * 16;             // A lane K sub-offset
  const int  kB   = hi * 16;             // B lane K sub-offset
  const bool ok1  = (n1 < Nact);
  const bool ok2  = (n2 < Nact);

  v8f acc1 = {}, acc2 = {};
  #pragma unroll
  for (int k0 = 0; k0 < K; k0 += 32) {
    const float* ap = A + mrow * (long)lda + k0 + kA;
    if (k0 + 32 < K) __builtin_prefetch(ap + 32, 0, 1);  // global_prefetch_b8
    v16h a;
    #pragma unroll
    for (int i = 0; i < 8; ++i) a[i]     = (_Float16)ap[i];
    #pragma unroll
    for (int i = 0; i < 8; ++i) a[8 + i] = (_Float16)ap[8 + i];

    v16h b1 = {}, b2 = {};
    if (ok1) {
      const float* wp = W + (long)n1 * ldw + k0 + kB;
      #pragma unroll
      for (int i = 0; i < 16; ++i) b1[i] = (_Float16)wp[i];
    }
    if (ok2) {
      const float* wp = W + (long)n2 * ldw + k0 + kB;
      #pragma unroll
      for (int i = 0; i < 16; ++i) b2[i] = (_Float16)wp[i];
    }
    acc1 = __builtin_amdgcn_wmma_f32_16x16x32_f16(false, a, false, b1,
                                                  (short)0, acc1, false, false);
    acc2 = __builtin_amdgcn_wmma_f32_16x16x32_f16(false, a, false, b2,
                                                  (short)0, acc2, false, false);
  }
  const long mb = m0 + 8 * hi;
  if (ok1) {
    #pragma unroll
    for (int r = 0; r < 8; ++r) D[(mb + r) * (long)ldd + n1] = acc1[r];
  }
  if (ok2) {
    #pragma unroll
    for (int r = 0; r < 8; ++r) D[(mb + r) * (long)ldd + n2] = acc2[r];
  }
}

// =====================================================================
// kd[o][c] = sum_{h,w} w[o][c][0][h][w] + w[o][c][2][h][w]
// =====================================================================
__global__ void k_compute_kd(const float* __restrict__ w, float* __restrict__ kd) {
  int i = blockIdx.x * blockDim.x + threadIdx.x;
  if (i >= CNUM * CNUM) return;
  const float* p = w + (long)i * 27;
  float s = 0.f;
  #pragma unroll
  for (int j = 0; j < 9; ++j) s += p[j] + p[18 + j];
  kd[i] = s;
}

// =====================================================================
// conv3d(3x3x3, pad 1) - theta * x . kd, then BN + ReLU; write xf[b][l][o]
// block = 64 threads (one out-channel each), grid = (LSEQ, B)
// =====================================================================
__global__ __launch_bounds__(64)
void k_conv_bn_relu(const float* __restrict__ x, const float* __restrict__ w,
                    const float* __restrict__ kd,
                    const float* __restrict__ bg, const float* __restrict__ bb,
                    const float* __restrict__ bm, const float* __restrict__ bv,
                    float* __restrict__ xf) {
  const int l = blockIdx.x;
  const int b = blockIdx.y;
  const int t = l >> 6, h = (l >> 3) & 7, wq = l & 7;
  const int o = threadIdx.x;
  __shared__ float xs[CNUM];
  float acc = 0.f;

  for (int kt = 0; kt < 3; ++kt) {
    int ts = t + kt - 1; if (ts < 0 || ts >= TNUM) continue;   // uniform
    for (int kh = 0; kh < 3; ++kh) {
      int hs = h + kh - 1; if (hs < 0 || hs >= HNUM) continue; // uniform
      for (int kw = 0; kw < 3; ++kw) {
        int wsr = wq + kw - 1; if (wsr < 0 || wsr >= WNUM) continue; // uniform
        __syncthreads();
        xs[o] = x[(((long)b * CNUM + o) * TNUM + ts) * 64 + hs * 8 + wsr];
        __syncthreads();
        const float* wp = w + ((long)o * CNUM) * 27 + (kt * 9 + kh * 3 + kw);
        float s = 0.f;
        for (int c = 0; c < CNUM; ++c) s += xs[c] * wp[c * 27];
        acc += s;
      }
    }
  }
  // temporal-difference term at center position
  __syncthreads();
  xs[o] = x[(((long)b * CNUM + o) * TNUM + t) * 64 + h * 8 + wq];
  __syncthreads();
  const float* kdp = kd + o * CNUM;
  float s2 = 0.f;
  for (int c = 0; c < CNUM; ++c) s2 += xs[c] * kdp[c];
  acc -= THETA * s2;

  float v = (acc - bm[o]) * rsqrtf(bv[o] + EPSV) * bg[o] + bb[o];
  v = fmaxf(v, 0.f);
  xf[((long)b * LSEQ + l) * CNUM + o] = v;
}

// =====================================================================
// LayerNorm over 64 channels.  grid = BL, block = 64.
// add3 != 0 -> normalize (x + y1 + y2)
// =====================================================================
__global__ __launch_bounds__(64)
void k_layernorm(const float* __restrict__ x, const float* __restrict__ y1,
                 const float* __restrict__ y2, const float* __restrict__ g,
                 const float* __restrict__ be, float* __restrict__ out, int add3) {
  const long row = blockIdx.x;
  const int c = threadIdx.x;
  float v = x[row * CNUM + c];
  if (add3) v += y1[row * CNUM + c] + y2[row * CNUM + c];
  __shared__ float red[CNUM];
  red[c] = v; __syncthreads();
  #pragma unroll
  for (int off = 32; off > 0; off >>= 1) {
    if (c < off) red[c] += red[c + off];
    __syncthreads();
  }
  const float mean = red[0] * (1.f / CNUM);
  __syncthreads();
  float dv = v - mean;
  red[c] = dv * dv; __syncthreads();
  #pragma unroll
  for (int off = 32; off > 0; off >>= 1) {
    if (c < off) red[c] += red[c + off];
    __syncthreads();
  }
  const float var = red[0] * (1.f / CNUM);
  out[row * CNUM + c] = dv * rsqrtf(var + EPSV) * g[c] + be[c];
}

// =====================================================================
// depthwise causal conv-4 over L + SiLU.  rev==1 -> anti-causal taps
// (implements the flip of the backward Mamba without materializing flips)
// =====================================================================
__global__ void k_dwconv_silu(const float* __restrict__ xz,
                              const float* __restrict__ cw,
                              const float* __restrict__ cb,
                              float* __restrict__ xc, int rev) {
  long gid = (long)blockIdx.x * blockDim.x + threadIdx.x;
  if (gid >= (long)BL * DI) return;
  const int d = (int)(gid & (DI - 1));
  const long bl = gid >> 7;
  const int l = (int)(bl % LSEQ);
  const int b = (int)(bl / LSEQ);
  float acc = cb[d];
  #pragma unroll
  for (int j = 0; j < 4; ++j) {
    int src = rev ? (l + 3 - j) : (l - 3 + j);
    if (src >= 0 && src < LSEQ)
      acc += cw[d * 4 + j] * xz[((long)b * LSEQ + src) * 256 + d];
  }
  xc[bl * DI + d] = acc * (1.f / (1.f + expf(-acc)));
}

// =====================================================================
// dt = softplus(dbl[:, :4] @ dt_w^T + dt_b)
// =====================================================================
__global__ void k_dtprep(const float* __restrict__ dbl,
                         const float* __restrict__ dtw,
                         const float* __restrict__ dtb,
                         float* __restrict__ dt) {
  long gid = (long)blockIdx.x * blockDim.x + threadIdx.x;
  if (gid >= (long)BL * DI) return;
  const int d = (int)(gid & (DI - 1));
  const long bl = gid >> 7;
  const float* p = dbl + bl * 36;
  float r = dtb[d];
  #pragma unroll
  for (int j = 0; j < DRN; ++j) r += p[j] * dtw[d * DRN + j];
  dt[gid] = (r > 20.f) ? r : log1pf(expf(r));
}

// =====================================================================
// Scan phase 1: per-chunk local scan (h from 0) + cumulative dA product.
// grid = (NCH, B), block = 256; thread t owns d = t>>1, s in [(t&1)*8, +8)
// =====================================================================
__global__ __launch_bounds__(256)
void k_scan_phase1(const float* __restrict__ dt, const float* __restrict__ xc,
                   const float* __restrict__ dbl, const float* __restrict__ alog,
                   float* __restrict__ hloc, float* __restrict__ Pc, int rev) {
  const int chunk = blockIdx.x, b = blockIdx.y;
  const int t = threadIdx.x;
  const int d = t >> 1, sb = (t & 1) * 8;
  float A[8], h[8], P[8];
  #pragma unroll
  for (int s = 0; s < 8; ++s) {
    A[s] = -expf(alog[d * DSN + sb + s]);
    h[s] = 0.f; P[s] = 1.f;
  }
  for (int i = 0; i < CHK; ++i) {
    const int m = chunk * CHK + i;
    const int l = rev ? (LSEQ - 1 - m) : m;
    const long base = (long)b * LSEQ + l;
    const float dtv = dt[base * DI + d];
    const float dx  = dtv * xc[base * DI + d];
    const float* Bp = dbl + base * 36 + DRN + sb;
    #pragma unroll
    for (int s = 0; s < 8; ++s) {
      float dA = expf(dtv * A[s]);
      h[s] = dA * h[s] + dx * Bp[s];
      P[s] *= dA;
    }
  }
  const long o = ((long)b * NCH + chunk) * PAIRS + d * DSN + sb;
  #pragma unroll
  for (int s = 0; s < 8; ++s) { hloc[o + s] = h[s]; Pc[o + s] = P[s]; }
}

// =====================================================================
// Scan phase 2: chunk-level prefix. 4096 independent (b,pair) recurrences.
// =====================================================================
__global__ void k_scan_phase2(const float* __restrict__ hloc,
                              const float* __restrict__ Pc,
                              float* __restrict__ Hin) {
  int gid = blockIdx.x * blockDim.x + threadIdx.x;
  if (gid >= BNUM * PAIRS) return;
  const int b = gid >> 11, p = gid & (PAIRS - 1);
  float H = 0.f;
  for (int k = 0; k < NCH; ++k) {
    const long idx = ((long)b * NCH + k) * PAIRS + p;
    Hin[idx] = H;
    H = hloc[idx] + Pc[idx] * H;
  }
}

// =====================================================================
// Scan phase 3: re-run chunk from true initial state, emit gated y.
// yg[b,l,d] = (sum_s h*Cm + D[d]*xc) * silu(z)
// =====================================================================
__global__ __launch_bounds__(256)
void k_scan_phase3(const float* __restrict__ dt, const float* __restrict__ xc,
                   const float* __restrict__ dbl, const float* __restrict__ alog,
                   const float* __restrict__ Dp, const float* __restrict__ xz,
                   const float* __restrict__ Hin, float* __restrict__ yg, int rev) {
  const int chunk = blockIdx.x, b = blockIdx.y;
  const int t = threadIdx.x;
  const int d = t >> 1, sb = (t & 1) * 8;
  float A[8], h[8];
  const long o = ((long)b * NCH + chunk) * PAIRS + d * DSN + sb;
  #pragma unroll
  for (int s = 0; s < 8; ++s) {
    A[s] = -expf(alog[d * DSN + sb + s]);
    h[s] = Hin[o + s];
  }
  for (int i = 0; i < CHK; ++i) {
    const int m = chunk * CHK + i;
    const int l = rev ? (LSEQ - 1 - m) : m;
    const long base = (long)b * LSEQ + l;
    const float dtv = dt[base * DI + d];
    const float xcv = xc[base * DI + d];
    const float dx  = dtv * xcv;
    const float* Bp = dbl + base * 36 + DRN + sb;
    const float* Cp = dbl + base * 36 + DRN + DSN + sb;
    float yp = 0.f;
    #pragma unroll
    for (int s = 0; s < 8; ++s) {
      float dA = expf(dtv * A[s]);
      h[s] = dA * h[s] + dx * Bp[s];
      yp += h[s] * Cp[s];
    }
    float yfull = yp + __shfl_xor(yp, 1, 32);   // combine s-halves (adjacent lanes)
    if ((t & 1) == 0) {
      float yv = yfull + Dp[d] * xcv;
      float zv = xz[base * 256 + DI + d];
      yg[base * DI + d] = yv * (zv * (1.f / (1.f + expf(-zv))));
    }
  }
}

// =====================================================================
// avg / max pool over L per (b,c).  grid = B*C, block = 256.
// =====================================================================
__global__ __launch_bounds__(256)
void k_pool(const float* __restrict__ xo, float* __restrict__ avg,
            float* __restrict__ mx) {
  const int bc = blockIdx.x;
  const int b = bc >> 6, c = bc & 63;
  const int t = threadIdx.x;
  float s = 0.f, m = -3.4e38f;
  for (int l = t; l < LSEQ; l += 256) {
    float v = xo[((long)b * LSEQ + l) * CNUM + c];
    s += v; m = fmaxf(m, v);
  }
  __shared__ float rs[256], rm[256];
  rs[t] = s; rm[t] = m; __syncthreads();
  #pragma unroll
  for (int off = 128; off > 0; off >>= 1) {
    if (t < off) { rs[t] += rs[t + off]; rm[t] = fmaxf(rm[t], rm[t + off]); }
    __syncthreads();
  }
  if (t == 0) { avg[bc] = rs[0] * (1.f / LSEQ); mx[bc] = rm[0]; }
}

// =====================================================================
// channel attention: sigmoid(fc(avg)+fc(mx)).  grid = B, block = 64.
// =====================================================================
__global__ __launch_bounds__(64)
void k_att(const float* __restrict__ avg, const float* __restrict__ mx,
           const float* __restrict__ w1, const float* __restrict__ w2,
           float* __restrict__ att) {
  const int b = blockIdx.x, t = threadIdx.x;
  __shared__ float sa[CNUM], sm[CNUM], ha[32], hm[32];
  sa[t] = avg[b * CNUM + t]; sm[t] = mx[b * CNUM + t];
  __syncthreads();
  if (t < 32) {
    float a0 = 0.f, m0 = 0.f;
    for (int c = 0; c < CNUM; ++c) {
      a0 += sa[c] * w1[t * CNUM + c];
      m0 += sm[c] * w1[t * CNUM + c];
    }
    ha[t] = fmaxf(a0, 0.f); hm[t] = fmaxf(m0, 0.f);
  }
  __syncthreads();
  float s = 0.f;
  for (int r = 0; r < 32; ++r) s += (ha[r] + hm[r]) * w2[t * 32 + r];
  att[b * CNUM + t] = 1.f / (1.f + expf(-s));
}

// =====================================================================
// out[b][c][t][h][w] = xo[b][l][c] * att[b][c]   (gid enumerates (b,c,l))
// =====================================================================
__global__ void k_final(const float* __restrict__ xo, const float* __restrict__ att,
                        float* __restrict__ out) {
  long gid = (long)blockIdx.x * blockDim.x + threadIdx.x;
  if (gid >= (long)BNUM * CNUM * LSEQ) return;
  const int l = (int)(gid % LSEQ);
  const long bc = gid / LSEQ;
  const int c = (int)(bc & 63), b = (int)(bc >> 6);
  out[gid] = xo[((long)b * LSEQ + l) * CNUM + c] * att[b * CNUM + c];
}

// =====================================================================
// host launcher
// =====================================================================
extern "C" void kernel_launch(void* const* d_in, const int* in_sizes, int n_in,
                              void* d_out, int out_size, void* d_ws, size_t ws_size,
                              hipStream_t stream) {
  const float* x     = (const float*)d_in[0];
  const float* tdcw  = (const float*)d_in[1];
  const float* bn_g  = (const float*)d_in[2];
  const float* bn_b  = (const float*)d_in[3];
  const float* bn_m  = (const float*)d_in[4];
  const float* bn_v  = (const float*)d_in[5];
  const float* ln1_g = (const float*)d_in[6];
  const float* ln1_b = (const float*)d_in[7];
  const float* ln2_g = (const float*)d_in[8];
  const float* ln2_b = (const float*)d_in[9];
  const float* ca_w1 = (const float*)d_in[10];
  const float* ca_w2 = (const float*)d_in[11];
  // mamba params: base 12 (forward), 21 (backward); order:
  // in_w, conv_w, conv_b, xproj_w, dt_w, dt_b, A_log, D, out_w

  float* ws = (float*)d_ws;
  size_t off = 0;
  float* xf     = ws + off; off += (size_t)BL * CNUM;
  float* xn     = ws + off; off += (size_t)BL * CNUM;
  float* xo     = ws + off; off += (size_t)BL * CNUM;
  float* yprojF = ws + off; off += (size_t)BL * CNUM;
  float* yprojB = ws + off; off += (size_t)BL * CNUM;
  float* kd     = ws + off; off += CNUM * CNUM;
  float* avg    = ws + off; off += BNUM * CNUM;
  float* mx     = ws + off; off += BNUM * CNUM;
  float* att    = ws + off; off += BNUM * CNUM;
  // direction-reused scratch
  float* xz     = ws + off; off += (size_t)BL * 256;
  float* xc     = ws + off; off += (size_t)BL * DI;
  float* dbl    = ws + off; off += (size_t)BL * 36;
  float* dtb    = ws + off; off += (size_t)BL * DI;
  float* hloc   = ws + off; off += (size_t)BNUM * NCH * PAIRS;
  float* Pc     = ws + off; off += (size_t)BNUM * NCH * PAIRS;
  float* Hin    = ws + off; off += (size_t)BNUM * NCH * PAIRS;
  float* yg     = ws + off; off += (size_t)BL * DI;

  // ---- stem: conv3d + TD + BN + ReLU, then LN1 ----
  k_compute_kd<<<(CNUM * CNUM + 255) / 256, 256, 0, stream>>>(tdcw, kd);
  k_conv_bn_relu<<<dim3(LSEQ, BNUM), 64, 0, stream>>>(x, tdcw, kd, bn_g, bn_b,
                                                      bn_m, bn_v, xf);
  k_layernorm<<<BL, 64, 0, stream>>>(xf, nullptr, nullptr, ln1_g, ln1_b, xn, 0);

  // ---- bidirectional mamba (flip handled via rev indexing) ----
  const long nElemDI = (long)BL * DI;
  for (int dir = 0; dir < 2; ++dir) {
    const int base = 12 + dir * 9;
    const float* in_w    = (const float*)d_in[base + 0];  // (256,64)
    const float* conv_w  = (const float*)d_in[base + 1];  // (128,1,4)
    const float* conv_b  = (const float*)d_in[base + 2];
    const float* xproj_w = (const float*)d_in[base + 3];  // (36,128)
    const float* dt_w    = (const float*)d_in[base + 4];  // (128,4)
    const float* dt_b    = (const float*)d_in[base + 5];
    const float* A_log   = (const float*)d_in[base + 6];  // (128,16)
    const float* Dvec    = (const float*)d_in[base + 7];
    const float* out_w   = (const float*)d_in[base + 8];  // (64,128)
    float* yproj = dir ? yprojB : yprojF;
    const int rev = dir;

    // in-proj: xz = xn @ in_w^T  (M=20480, N=256, K=64) -> 8 double-N-tiles
    wmma_gemm_nt<64><<<dim3(BL / 64, 8), 128, 0, stream>>>(
        xn, CNUM, in_w, CNUM, xz, 256, 256);
    // causal depthwise conv + SiLU
    k_dwconv_silu<<<(int)((nElemDI + 255) / 256), 256, 0, stream>>>(
        xz, conv_w, conv_b, xc, rev);
    // x-proj: dbl = xc @ xproj_w^T  (N=36 -> 2 double-tiles, K=128)
    wmma_gemm_nt<128><<<dim3(BL / 64, 2), 128, 0, stream>>>(
        xc, DI, xproj_w, DI, dbl, 36, 36);
    // dt = softplus(...)
    k_dtprep<<<(int)((nElemDI + 255) / 256), 256, 0, stream>>>(dbl, dt_w, dt_b, dtb);
    // chunked selective scan
    k_scan_phase1<<<dim3(NCH, BNUM), 256, 0, stream>>>(dtb, xc, dbl, A_log,
                                                       hloc, Pc, rev);
    k_scan_phase2<<<(BNUM * PAIRS + 255) / 256, 256, 0, stream>>>(hloc, Pc, Hin);
    k_scan_phase3<<<dim3(NCH, BNUM), 256, 0, stream>>>(dtb, xc, dbl, A_log, Dvec,
                                                       xz, Hin, yg, rev);
    // out-proj: yproj = yg @ out_w^T  (N=64 -> 2 double-tiles, K=128)
    wmma_gemm_nt<128><<<dim3(BL / 64, 2), 128, 0, stream>>>(
        yg, DI, out_w, DI, yproj, CNUM, CNUM);
  }

  // ---- residual + LN2, channel attention, final scale ----
  k_layernorm<<<BL, 64, 0, stream>>>(xf, yprojF, yprojB, ln2_g, ln2_b, xo, 1);
  k_pool<<<BNUM * CNUM, 256, 0, stream>>>(xo, avg, mx);
  k_att<<<BNUM, 64, 0, stream>>>(avg, mx, ca_w1, ca_w2, att);
  k_final<<<(int)(((long)BNUM * CNUM * LSEQ + 255) / 256), 256, 0, stream>>>(
      xo, att, (float*)d_out);
}
